// NDNGeneration_89704686944881
// MI455X (gfx1250) — matrix operand order repaired and checked by reading, not compiled, not run on hardware
//
#include <hip/hip_runtime.h>
#include <hip/hip_bf16.h>
#include <stdint.h>

// ---------------------------------------------------------------------------
// Model dims
//   L=128 N=9 E=64  O=1152 P=8192
//   genc: din {64,128,128}, h=512, dout=128
//   gu:   B=1152 batches of 9 nodes / 64 edges, din=128
// All matmuls are lrelu(A@W + b) -> one WMMA GEMM kernel with epilogue modes.
// ---------------------------------------------------------------------------

typedef __attribute__((ext_vector_type(16))) __bf16        v16bf;
typedef __attribute__((ext_vector_type(8)))  float         v8f;
typedef __attribute__((ext_vector_type(8)))  unsigned int  v8u;

static __device__ __forceinline__ unsigned int pk_bf16(float a, float b) {
  unsigned int ua = __builtin_bit_cast(unsigned int, a);
  unsigned int ub = __builtin_bit_cast(unsigned int, b);
  ua += 0x7FFFu + ((ua >> 16) & 1u);   // round-to-nearest-even
  ub += 0x7FFFu + ((ub >> 16) & 1u);
  return (ua >> 16) | (ub & 0xFFFF0000u);
}

// A-fragment mapping (16x32 bf16 per 16x16x32 wmma, wave32):
//   lane = (m&15) | (((k>>3)&1)<<4)
//   v    = ((k&16)>>2) | ((k>>1)&3),  half = k&1
// Packed array: dword index = frag*256 + lane*8 + v,  frag = mTile*KT + kTile
static __device__ __forceinline__ void a_coords(long tid, int KT, int& row, int& k) {
  int f = (int)(tid >> 8);
  int lv = (int)(tid & 255);
  int lane = lv >> 3, v = lv & 7;
  int mt = f / KT, kt = f - mt * KT;
  row = mt * 16 + (lane & 15);
  int kk = ((v & 4) << 2) | ((lane >> 4) << 3) | ((v & 3) << 1);
  k = kt * 32 + kk;
}

// ---------------------------- pack kernels ---------------------------------

__global__ void k_pack_a_plain(const float* __restrict__ A, unsigned int* __restrict__ out,
                               int KT, int K, long total) {
  long tid = (long)blockIdx.x * 256 + threadIdx.x;
  if (tid >= total) return;
  int row, k; a_coords(tid, KT, row, k);
  float x0 = (k     < K) ? A[(size_t)row * K + k    ] : 0.f;
  float x1 = (k + 1 < K) ? A[(size_t)row * K + k + 1] : 0.f;
  out[tid] = pk_bf16(x0, x1);
}

// B-fragment mapping (32x16, KxN): lane = (n&15) | (((k>>3)&1)<<4); v=((k&16)>>2)|((k>>1)&3)
__global__ void k_pack_b(const float* __restrict__ W, unsigned int* __restrict__ out,
                         int KT, int K, int N, long total) {
  long tid = (long)blockIdx.x * 256 + threadIdx.x;
  if (tid >= total) return;
  int f = (int)(tid >> 8);
  int lv = (int)(tid & 255);
  int lane = lv >> 3, v = lv & 7;
  int nt = f / KT, kt = f - nt * KT;
  int n = nt * 16 + (lane & 15);
  int k = kt * 32 + (((v & 4) << 2) | ((lane >> 4) << 3) | ((v & 3) << 1));
  float x0 = (k     < K && n < N) ? W[(size_t)k       * N + n] : 0.f;
  float x1 = (k + 1 < K && n < N) ? W[(size_t)(k + 1) * N + n] : 0.f;
  out[tid] = pk_bf16(x0, x1);
}

// genc concat gather:  X[r] = [obj[s_idx[r]] | pred[r] | obj[o_idx[r]]]
__global__ void k_pack_a_gather_genc(const float* __restrict__ obj, const float* __restrict__ pred,
                                     const int* __restrict__ si, const int* __restrict__ oi,
                                     unsigned int* __restrict__ out, int KT, int din, long total) {
  long tid = (long)blockIdx.x * 256 + threadIdx.x;
  if (tid >= total) return;
  int r, k; a_coords(tid, KT, r, k);
  auto fetch = [&](int kq) -> float {
    if (kq < din)       return obj [(size_t)si[r] * din + kq];
    if (kq < 2 * din)   return pred[(size_t)r     * din + (kq - din)];
    if (kq < 3 * din)   return obj [(size_t)oi[r] * din + (kq - 2 * din)];
    return 0.f;
  };
  out[tid] = pk_bf16(fetch(k), fetch(k + 1));
}

// gu concat gather: r = b*64+e, b = l*9+i;  uses local s/o (first 64 of s_idx/o_idx)
__global__ void k_pack_a_gather_gu(const float* __restrict__ OI, const float* __restrict__ pv,
                                   const int* __restrict__ si, const int* __restrict__ oi,
                                   unsigned int* __restrict__ out, int KT, long total) {
  long tid = (long)blockIdx.x * 256 + threadIdx.x;
  if (tid >= total) return;
  int r, k; a_coords(tid, KT, r, k);
  int b = r >> 6, e = r & 63, l = b / 9;
  size_t sRow = (size_t)(b * 9 + si[e]);
  size_t oRow = (size_t)(b * 9 + oi[e]);
  size_t pRow = (size_t)(l * 64 + e);
  auto fetch = [&](int kq) -> float {
    if (kq < 128) return OI[sRow * 128 + kq];
    if (kq < 256) return pv[pRow * 128 + (kq - 128)];
    if (kq < 384) return OI[oRow * 128 + (kq - 256)];
    return 0.f;
  };
  out[tid] = pk_bf16(fetch(k), fetch(k + 1));
}

// geometric input: r = (l*9+i)*9 + j ; [ov[l,j] (128) | boxes[l,j]*(j<i) (4) | pad]
__global__ void k_pack_a_geo(const float* __restrict__ ov, const float* __restrict__ boxes,
                             unsigned int* __restrict__ out, int KT, long total) {
  long tid = (long)blockIdx.x * 256 + threadIdx.x;
  if (tid >= total) return;
  int r, k; a_coords(tid, KT, r, k);
  int l = r / 81, rem = r - l * 81;
  int i = rem / 9, j = rem - i * 9;
  size_t orow = (size_t)(l * 9 + j);
  auto fetch = [&](int kq) -> float {
    if (kq < 128) return ov[orow * 128 + kq];
    if (kq < 132) return (j < i) ? boxes[orow * 4 + (kq - 128)] : 0.f;
    return 0.f;
  };
  out[tid] = pk_bf16(fetch(k), fetch(k + 1));
}

// two-tensor concat rows
__global__ void k_pack_a_cat2(const float* __restrict__ A, const float* __restrict__ B,
                              int wa, int wb, unsigned int* __restrict__ out, int KT, long total) {
  long tid = (long)blockIdx.x * 256 + threadIdx.x;
  if (tid >= total) return;
  int r, k; a_coords(tid, KT, r, k);
  auto fetch = [&](int kq) -> float {
    if (kq < wa)      return A[(size_t)r * wa + kq];
    if (kq < wa + wb) return B[(size_t)r * wb + (kq - wa)];
    return 0.f;
  };
  out[tid] = pk_bf16(fetch(k), fetch(k + 1));
}

// ------------------------------ GEMM ---------------------------------------
// C = lrelu(A @ W + b); each wave: 16(M) x 64(N) via 4 accumulators.
// mode 0: store;  mode 1: genc h2 scatter;  mode 2: gu h2 scatter.
__global__ void k_gemm(const unsigned int* __restrict__ Ap,
                       const unsigned int* __restrict__ Bp,
                       const float* __restrict__ bias,
                       float* __restrict__ out,
                       int MT, int KT, int Nact, int Nld, int mode,
                       const int* __restrict__ si, const int* __restrict__ oi,
                       float* __restrict__ aux, float* __restrict__ aux2) {
  const int lane = threadIdx.x & 31;
  const int wave = threadIdx.x >> 5;
  const int mTile = blockIdx.y * 8 + wave;
  if (mTile >= MT) return;               // wave-uniform; EXEC stays all-ones for wmma
  const int nt0 = blockIdx.x * 4;

  const v8u* a  = (const v8u*)(Ap + ((size_t)mTile * KT * 32 + lane) * 8);
  const v8u* b0 = (const v8u*)(Bp + ((size_t)(nt0 + 0) * KT * 32 + lane) * 8);
  const v8u* b1 = (const v8u*)(Bp + ((size_t)(nt0 + 1) * KT * 32 + lane) * 8);
  const v8u* b2 = (const v8u*)(Bp + ((size_t)(nt0 + 2) * KT * 32 + lane) * 8);
  const v8u* b3 = (const v8u*)(Bp + ((size_t)(nt0 + 3) * KT * 32 + lane) * 8);

  v8f acc0 = {}, acc1 = {}, acc2 = {}, acc3 = {};
  for (int kt = 0; kt < KT; ++kt) {
    v16bf av = __builtin_bit_cast(v16bf, a[(size_t)kt * 32]);
    acc0 = __builtin_amdgcn_wmma_f32_16x16x32_bf16(false, av, false,
             __builtin_bit_cast(v16bf, b0[(size_t)kt * 32]), (short)0, acc0, false, false);
    acc1 = __builtin_amdgcn_wmma_f32_16x16x32_bf16(false, av, false,
             __builtin_bit_cast(v16bf, b1[(size_t)kt * 32]), (short)0, acc1, false, false);
    acc2 = __builtin_amdgcn_wmma_f32_16x16x32_bf16(false, av, false,
             __builtin_bit_cast(v16bf, b2[(size_t)kt * 32]), (short)0, acc2, false, false);
    acc3 = __builtin_amdgcn_wmma_f32_16x16x32_bf16(false, av, false,
             __builtin_bit_cast(v16bf, b3[(size_t)kt * 32]), (short)0, acc3, false, false);
  }

  const int nLow = lane & 15;
  const int mRow = mTile * 16 + ((lane >> 4) << 3);   // C layout: lane -> n=lane&15, m=8*(lane>>4)+v
  auto epi = [&](v8f acc, int nTile) {
    int n0 = nTile * 16 + nLow;
    float bv = (bias && n0 < Nact) ? bias[n0] : 0.f;
#pragma unroll
    for (int v = 0; v < 8; ++v) {
      float x = acc[v] + bv;
      x = x > 0.f ? x : 0.2f * x;      // leaky relu follows every matmul in this model
      int m = mRow + v;
      if (mode == 0) {
        if (n0 < Nact) out[(size_t)m * Nld + n0] = x;
      } else if (mode == 1) {          // genc: new_s | new_p | new_o
        if (n0 < 512)       atomicAdd(aux + (size_t)si[m] * 512 + n0, x);
        else if (n0 < 640)  aux2[(size_t)m * 128 + (n0 - 512)] = x;
        else if (n0 < Nact) atomicAdd(aux + (size_t)oi[m] * 512 + (n0 - 640), x);
      } else {                         // gu: scatter into per-batch pooled / summed new_p
        int bb = m >> 6, e = m & 63;
        if (n0 < 512)       atomicAdd(aux + (size_t)(bb * 9 + si[e]) * 512 + n0, x);
        else if (n0 < 640)  atomicAdd(aux2 + (size_t)bb * 128 + (n0 - 512), x);
        else if (n0 < Nact) atomicAdd(aux + (size_t)(bb * 9 + oi[e]) * 512 + (n0 - 640), x);
      }
    }
  };
  epi(acc0, nt0); epi(acc1, nt0 + 1); epi(acc2, nt0 + 2); epi(acc3, nt0 + 3);
}

// ---------------------------- misc kernels ---------------------------------

__global__ void k_zero(float* p, long n) {
  long t = (long)blockIdx.x * 256 + threadIdx.x;
  if (t < n) p[t] = 0.f;
}

__global__ void k_count(float* __restrict__ cnt, const int* __restrict__ si,
                        const int* __restrict__ oi, int nE, int guMode) {
  int t = blockIdx.x * 256 + threadIdx.x;
  if (t >= nE) return;
  if (!guMode) {
    atomicAdd(cnt + si[t], 1.f);
    atomicAdd(cnt + oi[t], 1.f);
  } else {
    int b = t >> 6, e = t & 63;
    atomicAdd(cnt + b * 9 + si[e], 1.f);
    atomicAdd(cnt + b * 9 + oi[e], 1.f);
  }
}

__global__ void k_div_pooled(float* __restrict__ pooled, const float* __restrict__ cnt, long n) {
  long t = (long)blockIdx.x * 256 + threadIdx.x;
  if (t >= n) return;
  float d = cnt[t >> 9];
  pooled[t] /= (d > 1.f ? d : 1.f);
}

// c[b,f] = (sum_i uo[b*9+i, f] + upsum[b,f]) / (N+E)
__global__ void k_rowsum9(const float* __restrict__ uo, const float* __restrict__ upsum,
                          float* __restrict__ c, int n) {
  int t = blockIdx.x * 256 + threadIdx.x;
  if (t >= n) return;
  int b = t >> 7, f = t & 127;
  float s = upsum[t];
#pragma unroll
  for (int i = 0; i < 9; ++i) s += uo[(size_t)(b * 9 + i) * 128 + f];
  c[t] = s * (1.0f / 73.0f);
}

static __device__ __forceinline__ unsigned int pcg(unsigned int x) {
  x = x * 747796405u + 2891336453u;
  unsigned int w = ((x >> ((x >> 28) + 4u)) ^ x) * 277803737u;
  return (w >> 22) ^ w;
}

// z = eps * exp(0.5*var) + mu   (deterministic hash normal standing in for key-42)
__global__ void k_reparam(const float* __restrict__ mu, const float* __restrict__ var,
                          float* __restrict__ z, int n) {
  int t = blockIdx.x * 256 + threadIdx.x;
  if (t >= n) return;
  unsigned int h1 = pcg((unsigned)t * 2u + 12345u);
  unsigned int h2 = pcg((unsigned)t * 2u + 54321u);
  float u1 = ((float)h1 + 1.0f) * (1.0f / 4294967296.0f);
  float u2 = (float)h2 * (1.0f / 4294967296.0f);
  float eps = sqrtf(-2.0f * logf(u1)) * cosf(6.28318530718f * u2);
  z[t] = eps * expf(0.5f * var[t]) + mu[t];
}

// ------------------------------- host --------------------------------------

static inline long aTot(int M, int K) { return (long)(M / 16) * ((K + 31) / 32) * 256; }
static inline unsigned gsz(long total) { return (unsigned)((total + 255) / 256); }

extern "C" void kernel_launch(void* const* d_in, const int* in_sizes, int n_in,
                              void* d_out, int out_size, void* d_ws, size_t ws_size,
                              hipStream_t stream) {
  (void)in_sizes; (void)out_size; (void)ws_size;
  const float* obj_vecs = (const float*)d_in[1];   // (1152,64)
  const float* pred_vecs = (const float*)d_in[2];  // (8192,64)
  const float* boxes = (const float*)d_in[3];      // (1152,4)
  const int* s_idx = (const int*)d_in[4];          // (8192,)
  const int* o_idx = (const int*)d_in[5];          // (8192,)

  // params: pytree (sorted dict keys) flattening:
  // dec(3x W,b) enc_b enc_c enc_h enc_mu enc_var ge genc[3]{W1,W2,b1,b2,n2W1,n2W2,n2b1,n2b2} gu{...} pr(5x W,b)
  const float* Pm[60] = {nullptr};
  for (int i = 0; i < 60 && (6 + i) < n_in; ++i) Pm[i] = (const float*)d_in[6 + i];

  // workspace bump allocator
  char* wsp = (char*)d_ws;
  auto alloc = [&](size_t bytes) -> void* {
    void* p = wsp; wsp += (bytes + 255) & ~(size_t)255; return p;
  };
  unsigned int* wArena = (unsigned int*)alloc(16ull << 20);      // packed weights (bf16 frags)
  unsigned int* aArena = (unsigned int*)alloc(75497472ull);      // packed A (max 73728x512)
  float* bigbuf = (float*)alloc(150994944ull);                   // H1 / HU1 fp32
  float* pooled = (float*)alloc(21233664ull);                    // up to 10368x512
  float* p1buf  = (float*)alloc(21233664ull);
  float* uoBuf  = (float*)alloc(5308416ull);                     // 10368x128
  float* OIbuf  = (float*)alloc(5308416ull);
  float* ovA    = (float*)alloc(589824ull);
  float* ovB    = (float*)alloc(589824ull);
  float* pvA    = (float*)alloc(4194304ull);
  float* pvB    = (float*)alloc(4194304ull);
  float* upsum  = (float*)alloc(589824ull);
  float* cbuf   = (float*)alloc(589824ull);
  float* cnt    = (float*)alloc(41472ull);
  float* hb     = (float*)alloc(589824ull);
  float* hc     = (float*)alloc(589824ull);
  float* henc   = (float*)alloc(147456ull);
  float* zbuf   = (float*)alloc(147456ull);
  float* d1     = (float*)alloc(589824ull);
  float* d2     = (float*)alloc(294912ull);
  float* y1     = (float*)alloc(589824ull);
  float* y2     = (float*)alloc(589824ull);
  float* y3     = (float*)alloc(147456ull);

  // ---- pack all weights to bf16 fragment layout
  size_t wPos = 0;
  auto packW = [&](const float* W, int K, int N) -> const unsigned int* {
    int KT = (K + 31) / 32;
    int NTa = (((N + 15) / 16 + 3) / 4) * 4;
    long total = (long)NTa * KT * 256;
    unsigned int* dst = wArena + wPos;
    wPos += (size_t)total;
    k_pack_b<<<gsz(total), 256, 0, stream>>>(W, dst, KT, K, N, total);
    return dst;
  };
  const unsigned int* decW0p = packW(Pm[0], 160, 128);
  const unsigned int* decW1p = packW(Pm[2], 128, 64);
  const unsigned int* decW2p = packW(Pm[4], 64, 4);
  const unsigned int* encbWp = packW(Pm[6], 4, 128);
  const unsigned int* enccWp = packW(Pm[8], 128, 128);
  const unsigned int* enchWp = packW(Pm[10], 256, 32);
  const unsigned int* encmuWp = packW(Pm[12], 32, 32);
  const unsigned int* encvWp  = packW(Pm[14], 32, 32);
  const unsigned int* geWp = packW(Pm[16], 132, 128);
  const unsigned int *gW1p[3], *gW2p[3], *gn1p[3], *gn2p[3];
  const float *gb1[3], *gb2[3], *gnb1[3], *gnb2[3];
  for (int i = 0; i < 3; ++i) {
    const float* const* g = &Pm[18 + i * 8];  // W1 W2 b1 b2 n2W1 n2W2 n2b1 n2b2
    int din = (i == 0) ? 64 : 128;
    gW1p[i] = packW(g[0], 3 * din, 512);
    gW2p[i] = packW(g[1], 512, 1152);
    gn1p[i] = packW(g[4], 512, 512);
    gn2p[i] = packW(g[5], 512, 128);
    gb1[i] = g[2]; gb2[i] = g[3]; gnb1[i] = g[6]; gnb2[i] = g[7];
  }
  const float* const* gu = &Pm[42];
  const unsigned int* guW1p = packW(gu[0], 384, 512);
  const unsigned int* guW2p = packW(gu[1], 512, 1152);
  const unsigned int* gun1p = packW(gu[4], 512, 512);
  const unsigned int* gun2p = packW(gu[5], 512, 128);
  const unsigned int* pr0p = packW(Pm[50], 128, 128);
  const unsigned int* pr1p = packW(Pm[52], 128, 128);
  const unsigned int* pr2p = packW(Pm[54], 128, 32);
  const unsigned int* pr3p = packW(Pm[56], 32, 32);
  const unsigned int* pr4p = packW(Pm[58], 32, 32);

  auto gemm = [&](const unsigned int* Ap, const unsigned int* Bw, const float* bias, float* out,
                  int M, int K, int N, int Nld, int mode,
                  const int* si, const int* oi, float* aux, float* aux2) {
    int MT = M / 16, KT = (K + 31) / 32;
    int NTa = (((N + 15) / 16 + 3) / 4) * 4;
    dim3 grid((unsigned)(NTa / 4), (unsigned)((MT + 7) / 8));
    k_gemm<<<grid, 256, 0, stream>>>(Ap, Bw, bias, out, MT, KT, N, Nld, mode, si, oi, aux, aux2);
  };
  auto packAplain = [&](const float* A, int M, int K) {
    long total = aTot(M, K);
    k_pack_a_plain<<<gsz(total), 256, 0, stream>>>(A, aArena, (K + 31) / 32, K, total);
  };
  auto zero = [&](float* p, long n) { k_zero<<<gsz(n), 256, 0, stream>>>(p, n); };

  // ---------------- genc x3 ----------------
  const float* objIn = obj_vecs;
  const float* prIn = pred_vecs;
  float* ovOut[3] = {ovA, ovB, ovA};
  float* pvOut[3] = {pvA, pvB, pvA};
  for (int i = 0; i < 3; ++i) {
    int din = (i == 0) ? 64 : 128;
    long totX = aTot(8192, 3 * din);
    k_pack_a_gather_genc<<<gsz(totX), 256, 0, stream>>>(objIn, prIn, s_idx, o_idx,
                                                        aArena, (3 * din) / 32, din, totX);
    gemm(aArena, gW1p[i], gb1[i], bigbuf, 8192, 3 * din, 512, 512, 0, nullptr, nullptr, nullptr, nullptr);
    packAplain(bigbuf, 8192, 512);
    zero(pooled, 1152 * 512);
    gemm(aArena, gW2p[i], gb2[i], nullptr, 8192, 512, 1152, 0, 1, s_idx, o_idx, pooled, pvOut[i]);
    zero(cnt, 1152);
    k_count<<<gsz(8192), 256, 0, stream>>>(cnt, s_idx, o_idx, 8192, 0);
    k_div_pooled<<<gsz(1152L * 512), 256, 0, stream>>>(pooled, cnt, 1152L * 512);
    packAplain(pooled, 1152, 512);
    gemm(aArena, gn1p[i], gnb1[i], p1buf, 1152, 512, 512, 512, 0, nullptr, nullptr, nullptr, nullptr);
    packAplain(p1buf, 1152, 512);
    gemm(aArena, gn2p[i], gnb2[i], ovOut[i], 1152, 512, 128, 128, 0, nullptr, nullptr, nullptr, nullptr);
    objIn = ovOut[i]; prIn = pvOut[i];
  }

  // ---------------- geometric embed ----------------
  long totG = aTot(10368, 160);
  k_pack_a_geo<<<gsz(totG), 256, 0, stream>>>(objIn, boxes, aArena, 5, totG);
  gemm(aArena, geWp, Pm[17], OIbuf, 10368, 160, 128, 128, 0, nullptr, nullptr, nullptr, nullptr);

  // ---------------- gu gconv (dominant) ----------------
  long totU = aTot(73728, 384);
  k_pack_a_gather_gu<<<gsz(totU), 256, 0, stream>>>(OIbuf, prIn, s_idx, o_idx, aArena, 12, totU);
  gemm(aArena, guW1p, gu[2], bigbuf, 73728, 384, 512, 512, 0, nullptr, nullptr, nullptr, nullptr);
  packAplain(bigbuf, 73728, 512);
  zero(pooled, 10368L * 512);
  zero(upsum, 1152L * 128);
  gemm(aArena, guW2p, gu[3], nullptr, 73728, 512, 1152, 0, 2, s_idx, o_idx, pooled, upsum);
  zero(cnt, 10368);
  k_count<<<gsz(73728), 256, 0, stream>>>(cnt, s_idx, o_idx, 73728, 1);
  k_div_pooled<<<gsz(10368L * 512), 256, 0, stream>>>(pooled, cnt, 10368L * 512);
  packAplain(pooled, 10368, 512);
  gemm(aArena, gun1p, gu[6], p1buf, 10368, 512, 512, 512, 0, nullptr, nullptr, nullptr, nullptr);
  packAplain(p1buf, 10368, 512);
  gemm(aArena, gun2p, gu[7], uoBuf, 10368, 512, 128, 128, 0, nullptr, nullptr, nullptr, nullptr);
  k_rowsum9<<<gsz(1152L * 128), 256, 0, stream>>>(uoBuf, upsum, cbuf, 1152 * 128);

  // ---------------- encoder heads ----------------
  float* outF = (float*)d_out;  // [pred_boxes 4608][mu 36864][var 36864][mu_p 36864][var_p 36864]
  packAplain(boxes, 1152, 4);
  gemm(aArena, encbWp, Pm[7], hb, 1152, 4, 128, 128, 0, nullptr, nullptr, nullptr, nullptr);
  packAplain(cbuf, 1152, 128);
  gemm(aArena, enccWp, Pm[9], hc, 1152, 128, 128, 128, 0, nullptr, nullptr, nullptr, nullptr);
  long totH = aTot(1152, 256);
  k_pack_a_cat2<<<gsz(totH), 256, 0, stream>>>(hb, hc, 128, 128, aArena, 8, totH);
  gemm(aArena, enchWp, Pm[11], henc, 1152, 256, 32, 32, 0, nullptr, nullptr, nullptr, nullptr);
  packAplain(henc, 1152, 32);
  gemm(aArena, encmuWp, Pm[13], outF + 4608, 1152, 32, 32, 32, 0, nullptr, nullptr, nullptr, nullptr);
  gemm(aArena, encvWp,  Pm[15], outF + 41472, 1152, 32, 32, 32, 0, nullptr, nullptr, nullptr, nullptr);
  k_reparam<<<gsz(36864), 256, 0, stream>>>(outF + 4608, outF + 41472, zbuf, 36864);

  // ---------------- decoder ----------------
  long totD = aTot(1152, 160);
  k_pack_a_cat2<<<gsz(totD), 256, 0, stream>>>(zbuf, cbuf, 32, 128, aArena, 5, totD);
  gemm(aArena, decW0p, Pm[1], d1, 1152, 160, 128, 128, 0, nullptr, nullptr, nullptr, nullptr);
  packAplain(d1, 1152, 128);
  gemm(aArena, decW1p, Pm[3], d2, 1152, 128, 64, 64, 0, nullptr, nullptr, nullptr, nullptr);
  packAplain(d2, 1152, 64);
  gemm(aArena, decW2p, Pm[5], outF, 1152, 64, 4, 4, 0, nullptr, nullptr, nullptr, nullptr);

  // ---------------- prior heads ----------------
  packAplain(cbuf, 1152, 128);
  gemm(aArena, pr0p, Pm[51], y1, 1152, 128, 128, 128, 0, nullptr, nullptr, nullptr, nullptr);
  packAplain(y1, 1152, 128);
  gemm(aArena, pr1p, Pm[53], y2, 1152, 128, 128, 128, 0, nullptr, nullptr, nullptr, nullptr);
  packAplain(y2, 1152, 128);
  gemm(aArena, pr2p, Pm[55], y3, 1152, 128, 32, 32, 0, nullptr, nullptr, nullptr, nullptr);
  packAplain(y3, 1152, 32);
  gemm(aArena, pr3p, Pm[57], outF + 78336, 1152, 32, 32, 32, 0, nullptr, nullptr, nullptr, nullptr);
  gemm(aArena, pr4p, Pm[59], outF + 115200, 1152, 32, 32, 32, 0, nullptr, nullptr, nullptr, nullptr);
}
